// VaritationGCNEncoder_85237920956850
// MI455X (gfx1250) — matrix operand order repaired and checked by reading, not compiled, not run on hardware
//
#include <hip/hip_runtime.h>

typedef __attribute__((ext_vector_type(16))) _Float16 v16h;
typedef __attribute__((ext_vector_type(8)))  float    v8f;

#define N_NODES 50000

// Guaranteed-hardware fp32 atomic add (no return -> STOREcnt-tracked L2 RMW).
__device__ __forceinline__ void hw_atomic_add_f32(float* p, float v) {
    asm volatile("global_atomic_add_f32 %0, %1, off" :: "v"(p), "v"(v) : "memory");
}

// ---------------- init: dinv = 1 (self-loop), zero agg1 buffer ----------------
__global__ void k_init(float* __restrict__ dinv, float* __restrict__ bufA, long n) {
    long i = (long)blockIdx.x * blockDim.x + threadIdx.x;
    if (i < n) bufA[i] = 0.0f;
    if (i < N_NODES) dinv[i] = 1.0f;
}

// ---------------- degree accumulation over dst ----------------
__global__ void k_degree(const int* __restrict__ dstE, float* __restrict__ dinv, int n_edges) {
    int e = blockIdx.x * blockDim.x + threadIdx.x;
    if (e < n_edges) hw_atomic_add_f32(&dinv[dstE[e]], 1.0f);
}

// ---------------- dinv = rsqrt(deg)  (deg >= 1 always due to self loops) ----------------
__global__ void k_rsqrt(float* __restrict__ dinv) {
    int i = blockIdx.x * blockDim.x + threadIdx.x;
    if (i < N_NODES) dinv[i] = rsqrtf(dinv[i]);
}

// ---------------- WMMA GEMM: OUT = A[50000x128] * B[128x128] ----------------
// mode 0: B = B0 (128x128 row-major), plain store to out[row*128+col]
// mode 1: B = [B0 | B1] (each 128x64), add bias, split store:
//         col<64 -> out[row*64+col] (mu) ; col>=64 -> out[N*64 + row*64 + col-64] (logstd)
__global__ __launch_bounds__(128) void k_gemm(
    const float* __restrict__ A, const float* __restrict__ B0,
    const float* __restrict__ B1, const float* __restrict__ bias0,
    const float* __restrict__ bias1, float* __restrict__ out, int mode)
{
    // B staged in WMMA fragment layout: [kt][nt][lane][halfidx], 32 KB LDS
    __shared__ __align__(32) _Float16 wfrag[4][8][32][16];

    const int tid = threadIdx.x;
    for (int idx = tid; idx < 128 * 128; idx += 128) {
        int K = idx >> 7, N = idx & 127;
        float w;
        if (B1 == nullptr) w = B0[K * 128 + N];
        else               w = (N < 64) ? B0[K * 64 + N] : B1[K * 64 + (N - 64)];
        int kt = K >> 5, kin = K & 31, nt = N >> 4, nin = N & 15;
        int l = nin + (((kin >> 3) & 1) << 4);   // lanes 0-15: K 0-7,16-23 ; lanes 16-31: K 8-15,24-31
        int j = (kin & 7) + ((kin >> 4) << 3);
        wfrag[kt][nt][l][j] = (_Float16)w;
    }
    __syncthreads();

    const int lane  = tid & 31;
    const int wave  = tid >> 5;
    const int mrow  = lane & 15;           // A row within tile held by this lane
    const int hi8   = (lane >> 4) * 8;     // K-group base: 0 (lanes 0-15) or 8 (lanes 16-31)
    const int nwaves = gridDim.x * 4;
    const int ntiles = N_NODES / 16;       // 3125, exact

    for (int tile = blockIdx.x * 4 + wave; tile < ntiles; tile += nwaves) {
        const float* arow = A + (long)(tile * 16 + mrow) * 128;
        v16h afrag[4];
#pragma unroll
        for (int kt = 0; kt < 4; ++kt) {
            const float* p = arow + kt * 32 + hi8;   // lane holds K = base+{0..7, 16..23}
#pragma unroll
            for (int j = 0; j < 8; ++j) {
                afrag[kt][j]     = (_Float16)p[j];
                afrag[kt][j + 8] = (_Float16)p[j + 16];
            }
        }
#pragma unroll
        for (int nt = 0; nt < 8; ++nt) {
            v8f c = {};
#pragma unroll
            for (int kt = 0; kt < 4; ++kt) {
                v16h b = *(const v16h*)&wfrag[kt][nt][lane][0];
                c = __builtin_amdgcn_wmma_f32_16x16x32_f16(
                        false, afrag[kt], false, b, (short)0, c, false, false);
            }
            // C/D layout: VGPR r -> M = r + 8*(lane>=16), N = nt*16 + (lane&15)
            const int colbase = nt * 16 + (lane & 15);
            const int rowoff  = (lane >> 4) * 8;
            const long row0   = (long)tile * 16 + rowoff;
            if (mode == 0) {
                float* po = out + row0 * 128 + colbase;
#pragma unroll
                for (int r = 0; r < 8; ++r) po[r * 128] = c[r];
            } else {
                float bv = (colbase < 64) ? bias0[colbase] : bias1[colbase - 64];
                float* po = (colbase < 64)
                    ? out + row0 * 64 + colbase
                    : out + (long)N_NODES * 64 + row0 * 64 + (colbase - 64);
#pragma unroll
                for (int r = 0; r < 8; ++r) po[r * 64] = c[r] + bv;
            }
        }
    }
}

// ---------------- edge aggregation: agg[dst] += h[src] * dinv[src]*dinv[dst] ----------------
// one wave (32 lanes) per edge; lane handles 4 consecutive features (float4), 128 total.
// virtual edges >= n_edges are the self-loops. Edge id is wave-uniform -> scalarized loads.
__global__ __launch_bounds__(256) void k_agg(
    const float* __restrict__ hsrc, float* __restrict__ agg,
    const int* __restrict__ srcE, const int* __restrict__ dstE,
    const float* __restrict__ dinv, int n_edges)
{
    const int lane  = threadIdx.x & 31;
    int e = __builtin_amdgcn_readfirstlane(blockIdx.x * 8 + (threadIdx.x >> 5));
    const int total = n_edges + N_NODES;
    if (e >= total) return;
    int s, d;
    if (e < n_edges) { s = srcE[e]; d = dstE[e]; }
    else             { s = d = e - n_edges; }
    const float nrm = dinv[s] * dinv[d];
    const float4 v = *(const float4*)(hsrc + (long)s * 128 + lane * 4);
    float* o = agg + (long)d * 128 + lane * 4;
    asm volatile(
        "global_atomic_add_f32 %0, %1, off\n\t"
        "global_atomic_add_f32 %0, %2, off offset:4\n\t"
        "global_atomic_add_f32 %0, %3, off offset:8\n\t"
        "global_atomic_add_f32 %0, %4, off offset:12"
        :: "v"(o), "v"(v.x * nrm), "v"(v.y * nrm), "v"(v.z * nrm), "v"(v.w * nrm)
        : "memory");
}

// ---------------- h = relu(agg1 + b1) in place; zero bufB for next scatter ----------------
__global__ void k_bias_relu_zero(float* __restrict__ bufA, float* __restrict__ bufB,
                                 const float* __restrict__ b1, long n) {
    long i = (long)blockIdx.x * blockDim.x + threadIdx.x;
    if (i < n) {
        float v = bufA[i] + b1[i & 127];
        bufA[i] = v > 0.0f ? v : 0.0f;
        bufB[i] = 0.0f;
    }
}

extern "C" void kernel_launch(void* const* d_in, const int* in_sizes, int n_in,
                              void* d_out, int out_size, void* d_ws, size_t ws_size,
                              hipStream_t stream) {
    const float* x   = (const float*)d_in[0];
    const int*  edge = (const int*)  d_in[1];
    const float* W1  = (const float*)d_in[2];
    const float* b1  = (const float*)d_in[3];
    const float* Wmu = (const float*)d_in[4];
    const float* bmu = (const float*)d_in[5];
    const float* Wls = (const float*)d_in[6];
    const float* bls = (const float*)d_in[7];
    (void)n_in; (void)out_size; (void)ws_size;

    const int n_edges = in_sizes[1] / 2;
    const int* srcE = edge;              // edge_index[0]
    const int* dstE = edge + n_edges;    // edge_index[1]

    float* ws   = (float*)d_ws;
    float* dinv = ws;                              // 50000 (padded to 65536)
    float* bufA = ws + 65536;                      // 50000*128 = 6.4M
    float* bufB = bufA + (long)N_NODES * 128;      // 6.4M
    float* out  = (float*)d_out;                   // mu (50000*64) ++ logstd (50000*64)

    const long nfeat = (long)N_NODES * 128;
    const int  B = 256;
    const int  nagg_blocks = (n_edges + N_NODES + 7) / 8;   // one wave per (edge|self-loop)

    // 1) init: dinv=1 (self loop), zero agg1
    k_init<<<(int)((nfeat + B - 1) / B), B, 0, stream>>>(dinv, bufA, nfeat);
    // 2) in-degree of dst
    k_degree<<<(n_edges + B - 1) / B, B, 0, stream>>>(dstE, dinv, n_edges);
    // 3) dinv = rsqrt(deg)
    k_rsqrt<<<(N_NODES + B - 1) / B, B, 0, stream>>>(dinv);
    // 4) bufB = x @ W1            (WMMA)
    k_gemm<<<256, 128, 0, stream>>>(x, W1, nullptr, nullptr, nullptr, bufB, 0);
    // 5) bufA[dst] += bufB[src] * norm   (incl. self loops)
    k_agg<<<nagg_blocks, B, 0, stream>>>(bufB, bufA, srcE, dstE, dinv, n_edges);
    // 6) h = relu(bufA + b1) in place; zero bufB for reuse as agg2
    k_bias_relu_zero<<<(int)((nfeat + B - 1) / B), B, 0, stream>>>(bufA, bufB, b1, nfeat);
    // 7) bufB[dst] += h[src] * norm      (A·h, shared by mu and logstd)
    k_agg<<<nagg_blocks, B, 0, stream>>>(bufA, bufB, srcE, dstE, dinv, n_edges);
    // 8) out = (A·h) @ [Wmu|Wls] + [bmu|bls], split into mu / logstd   (WMMA)
    k_gemm<<<256, 128, 0, stream>>>(bufB, Wmu, Wls, bmu, bls, out, 1);
}
